// TokenizationLayer_15161234554927
// MI455X (gfx1250) — 8-bit matrix operand order repaired and checked by reading, compiled, not run on hardware
//
#include <hip/hip_runtime.h>


typedef __attribute__((ext_vector_type(16))) int   v16i;
typedef __attribute__((ext_vector_type(8)))  float v8f;

#define NC    30
#define BB    32
#define SS    2048
#define NN    512
#define PP    13
#define SCONV 2036

#define FP8_P1 0x38u
#define FP8_M1 0xB8u

#define WS_STREAM 0
#define WS_PCHAR  65536
#define WS_BSW    73728

__global__ __launch_bounds__(256) void decode_stream_k(
    const float* __restrict__ inp, unsigned char* __restrict__ stream)
{
  int t = blockIdx.x * 256 + threadIdx.x;
  int b = t >> 11, s = t & (SS - 1);
  const float* p = inp + (size_t)b * NC * SS + s;
  float acc = 0.0f;
#pragma unroll
  for (int c = 0; c < NC; ++c) acc += p[(size_t)c * SS] * (float)c;
  stream[t] = (unsigned char)(int)(acc + 0.5f);
}

__global__ __launch_bounds__(256) void decode_patterns_k(
    const float* __restrict__ pat, unsigned char* __restrict__ pchar)
{
  int t = blockIdx.x * 256 + threadIdx.x;
  if (t >= PP * NN) return;
  int p = t >> 9, n = t & (NN - 1);
  float best = -1.0f; int bc = 0;
#pragma unroll
  for (int c = 0; c < NC; ++c) {
    float v = pat[(((size_t)c * PP + p) << 9) + n];
    if (v > best) { best = v; bc = c; }
  }
  pchar[t] = (unsigned char)bc;
}

__global__ __launch_bounds__(256) void build_B_k(
    const unsigned char* __restrict__ pchar, unsigned* __restrict__ Bsw)
{
  int t = blockIdx.x * 256 + threadIdx.x;
  int lane = t & 31;
  int n = ((t >> 5) << 4) + (lane & 15);
  int h = lane >> 4;

  unsigned long long lo = 0ull; unsigned hi = 0u;
#pragma unroll
  for (int p = 0; p < PP; ++p) {
    unsigned long long w = pchar[p * NN + n];
    if (p < 12) lo |= w << (5 * p);
    else { lo |= (w & 0xFull) << 60; hi = (unsigned)(w >> 4); }
  }
#pragma unroll
  for (int v = 0; v < 16; ++v) {
    int k0 = ((v >> 2) << 5) + (h << 4) + ((v & 3) << 2);
    unsigned d = 0u;
#pragma unroll
    for (int j = 0; j < 4; ++j) {
      int k = k0 + j;
      unsigned bit  = (k < 64) ? ((unsigned)(lo >> (k & 63)) & 1u) : (hi & 1u);
      unsigned byte = (k < 65) ? (bit ? FP8_P1 : FP8_M1) : 0u;
      d |= byte << (8 * j);
    }
    Bsw[t * 16 + v] = d;
  }
}

__global__ __launch_bounds__(256) void tok_match_wmma_k(
    const unsigned char* __restrict__ stream,
    const uint4* __restrict__ Bsw,
    float* __restrict__ out)
{
  __shared__ uint4 Blds[4096];
  for (int i = threadIdx.x; i < 4096; i += 256) Blds[i] = Bsw[i];
  __syncthreads();

  const int wave = blockIdx.x * 8 + (threadIdx.x >> 5);
  const int lane = threadIdx.x & 31;
  const int b  = wave >> 7;
  const int s0 = (wave & 127) << 4;
  const int M  = lane & 15;
  const int g  = lane >> 4;

  const unsigned char* sb = stream + b * SS;
  unsigned long long lo = 0ull; unsigned hi = 0u;
#pragma unroll
  for (int p = 0; p < PP; ++p) {
    int s = s0 + M + p; if (s > SS - 1) s = SS - 1;
    unsigned long long w = sb[s];
    if (p < 12) lo |= w << (5 * p);
    else { lo |= (w & 0xFull) << 60; hi = (unsigned)(w >> 4); }
  }

  v16i A;
#pragma unroll
  for (int v = 0; v < 16; ++v) {
    int k0 = ((v >> 2) << 5) + (g << 4) + ((v & 3) << 2);
    unsigned d = 0u;
#pragma unroll
    for (int j = 0; j < 4; ++j) {
      int k = k0 + j;
      unsigned bit  = (k < 64) ? ((unsigned)(lo >> (k & 63)) & 1u) : (hi & 1u);
      unsigned byte = (k < 65) ? (bit ? FP8_P1 : FP8_M1) : 0u;
      d |= byte << (8 * j);
    }
    A[v] = (int)d;
  }

  const int   sLane = s0 + (g << 3);
  float* const outB = out + (size_t)b * NN * SS;

  for (int nt = 0; nt < 32; ++nt) {
    const uint4* bt = &Blds[(nt * 32 + lane) * 4];
    uint4 q0 = bt[0], q1 = bt[1], q2 = bt[2], q3 = bt[3];
    v16i Bv;
    Bv[0]=(int)q0.x; Bv[1]=(int)q0.y; Bv[2]=(int)q0.z;  Bv[3]=(int)q0.w;
    Bv[4]=(int)q1.x; Bv[5]=(int)q1.y; Bv[6]=(int)q1.z;  Bv[7]=(int)q1.w;
    Bv[8]=(int)q2.x; Bv[9]=(int)q2.y; Bv[10]=(int)q2.z; Bv[11]=(int)q2.w;
    Bv[12]=(int)q3.x;Bv[13]=(int)q3.y;Bv[14]=(int)q3.z; Bv[15]=(int)q3.w;

    v8f c = {};
    c = __builtin_amdgcn_wmma_f32_16x16x128_fp8_fp8(A, Bv, (short)0, c,
                                                     false,  false);

    float r0 = ((c[0] == 65.0f) && (sLane + 0 < SCONV)) ? 1.0f : 0.0f;
    float r1 = ((c[1] == 65.0f) && (sLane + 1 < SCONV)) ? 1.0f : 0.0f;
    float r2 = ((c[2] == 65.0f) && (sLane + 2 < SCONV)) ? 1.0f : 0.0f;
    float r3 = ((c[3] == 65.0f) && (sLane + 3 < SCONV)) ? 1.0f : 0.0f;
    float r4 = ((c[4] == 65.0f) && (sLane + 4 < SCONV)) ? 1.0f : 0.0f;
    float r5 = ((c[5] == 65.0f) && (sLane + 5 < SCONV)) ? 1.0f : 0.0f;
    float r6 = ((c[6] == 65.0f) && (sLane + 6 < SCONV)) ? 1.0f : 0.0f;
    float r7 = ((c[7] == 65.0f) && (sLane + 7 < SCONV)) ? 1.0f : 0.0f;

    const int n = (nt << 4) + M;
    float* op = outB + (size_t)n * SS + sLane;
    *(float4*)(op)     = make_float4(r0, r1, r2, r3);
    *(float4*)(op + 4) = make_float4(r4, r5, r6, r7);
  }
}

extern "C" void kernel_launch(void* const* d_in, const int* in_sizes, int n_in,
                              void* d_out, int out_size, void* d_ws, size_t ws_size,
                              hipStream_t stream)
{
  const float* input    = (const float*)d_in[0];
  const float* patterns = (const float*)d_in[1];
  float* out = (float*)d_out;

  unsigned char* ws     = (unsigned char*)d_ws;
  unsigned char* streamC= ws + WS_STREAM;
  unsigned char* pchar  = ws + WS_PCHAR;
  unsigned*      BswU   = (unsigned*)(ws + WS_BSW);
  const uint4*   Bsw4   = (const uint4*)(ws + WS_BSW);

  hipLaunchKernelGGL(decode_stream_k,   dim3(256), dim3(256), 0, stream, input, streamC);
  hipLaunchKernelGGL(decode_patterns_k, dim3(26),  dim3(256), 0, stream, patterns, pchar);
  hipLaunchKernelGGL(build_B_k,         dim3(4),   dim3(256), 0, stream, pchar, BswU);
  hipLaunchKernelGGL(tok_match_wmma_k,  dim3(512), dim3(256), 0, stream, streamC, Bsw4, out);
}
